// TokenDownLayer_6030134083851
// MI455X (gfx1250) — compile-verified, matrix operand order checked by reading, not run on hardware
//
#include <hip/hip_runtime.h>
#include <hip/hip_bf16.h>

// ---------------- problem constants (match reference) ----------------
#define BB 8
#define NN 12288
#define NTOK 3072
#define CC 256
#define COUT 512
#define HH 128
#define WW 96
#define H2 64
#define W2 48
#define HW (HH * WW)        // 12288
#define H2W2 (H2 * W2)      // 3072
#define MROWS (BB * NTOK)   // 24576 GEMM rows

// ---------------- workspace layout (float offsets) ----------------
// zeroed zone (accumulators):
#define OFF_MAPSUM 0ULL                               // B*HW*C        = 25165824
#define OFF_MAPCNT 25165824ULL                        // B*HW          =    98304
#define OFF_DD     25264128ULL                        // B*NT*C        =  6291456
#define OFF_M2T    31555584ULL                        // B*NT*C        =  6291456
#define OFF_DEN    37847040ULL                        // B*NT          =    24576
#define OFF_BN1S   37871616ULL                        // 256
#define OFF_BN1Q   37871872ULL                        // 256
#define OFF_BN2S   37872128ULL                        // 512
#define OFF_BN2Q   37872640ULL                        // 512
#define ZTOT       37873152ULL                        // floats to zero
// non-zeroed:
#define OFF_FMAP2  37873152ULL                        // B*H2W2*C      =  6291456
#define OFF_WBF    44164608ULL                        // CO*C/2 dwords =    65536
#define OFF_S1     44230144ULL
#define OFF_B1     44230400ULL
#define OFF_S2     44230656ULL
#define OFF_B2     44231168ULL
// aliases inside dead MAPSUM region (safe: MAPSUM last read in conv):
#define OFF_XD     0ULL                               //  6291456
#define OFF_Y      6291456ULL                         // 12582912
#define OFF_XDBF   18874368ULL                        //  3145728 dwords

typedef __attribute__((ext_vector_type(16))) __bf16 v16bf;
typedef __attribute__((ext_vector_type(8)))  float  v8f;

__device__ __forceinline__ unsigned f2bf_rne(float f) {
    unsigned u = __float_as_uint(f);
    u += 0x7FFFu + ((u >> 16) & 1u);
    return u >> 16;
}
__device__ __forceinline__ unsigned pack_bf16(float a, float b) {
    return f2bf_rne(a) | (f2bf_rne(b) << 16);
}
__device__ __forceinline__ int grid_idx(float lx, float ly, int h, int w) {
    lx = fminf(fmaxf(lx, -1.f), 1.f);
    ly = fminf(fmaxf(ly, -1.f), 1.f);
    int xi = (int)rintf(0.5f * (lx + 1.f) * (float)w - 0.5f);
    int yi = (int)rintf(0.5f * (ly + 1.f) * (float)h - 0.5f);
    xi = min(max(xi, 0), w - 1);
    yi = min(max(yi, 0), h - 1);
    return yi * w + xi;
}

// -------- zero the accumulator zone (float4 grid-stride) --------
__global__ void k_zero(float4* p, size_t n4) {
    size_t i = (size_t)blockIdx.x * blockDim.x + threadIdx.x;
    size_t s = (size_t)gridDim.x * blockDim.x;
    float4 z = make_float4(0.f, 0.f, 0.f, 0.f);
    for (; i < n4; i += s) p[i] = z;
}

// -------- convert GEMM weights (CO x C, row-major) to packed bf16 --------
__global__ void k_convert_w(const float* __restrict__ w, unsigned* __restrict__ wbf) {
    int i = blockIdx.x * blockDim.x + threadIdx.x;   // dword index, CO*C/2 total
    wbf[i] = pack_bf16(w[2 * i], w[2 * i + 1]);
}

// -------- fused token2map scatter + token_downup scatter (x read once) --------
__global__ void k_scatter1(const float* __restrict__ x, const float* __restrict__ loc,
                           const int* __restrict__ idx_agg, const int* __restrict__ idx_agg_t,
                           const float* __restrict__ wgt_t,
                           float* __restrict__ map_sum, float* __restrict__ map_cnt,
                           float* __restrict__ dd, float* __restrict__ den) {
    int token = blockIdx.x;                 // B*N blocks
    int b = token / NN;
    int c = threadIdx.x;                    // 256 channels
    int seg = b * HW + grid_idx(loc[2 * token], loc[2 * token + 1], HH, WW);
    int src = idx_agg[token];
    float val = x[((size_t)b * NN + src) * CC + c];
    atomicAdd(&map_sum[(size_t)seg * CC + c], val);
    int t = idx_agg_t[token];
    float wg = wgt_t[token];
    atomicAdd(&dd[((size_t)b * NTOK + t) * CC + c], val * wg);
    if (c == 0) {
        atomicAdd(&map_cnt[seg], 1.f);
        atomicAdd(&den[b * NTOK + t], wg);
    }
}

// -------- grid normalize + depthwise 3x3 stride-2 conv --------
__global__ void k_dwconv(const float* __restrict__ map_sum, const float* __restrict__ map_cnt,
                         const float* __restrict__ wconv, float* __restrict__ fmap2) {
    int p = blockIdx.x;                     // B*H2W2 output pixels
    int b = p / H2W2;
    int rem = p % H2W2;
    int h2 = rem / W2, w2 = rem % W2;
    int c = threadIdx.x;
    float acc = 0.f;
#pragma unroll
    for (int ky = 0; ky < 3; ++ky) {
        int ih = 2 * h2 - 1 + ky;
        if (ih < 0 || ih >= HH) continue;
#pragma unroll
        for (int kx = 0; kx < 3; ++kx) {
            int iw = 2 * w2 - 1 + kx;
            if (iw < 0 || iw >= WW) continue;
            int hw = b * HW + ih * WW + iw;
            float s = map_sum[(size_t)hw * CC + c];
            float cnt = map_cnt[hw];
            acc += wconv[c * 9 + ky * 3 + kx] * (s / (cnt + 1e-6f));
        }
    }
    fmap2[((size_t)b * H2W2 + h2 * W2 + w2) * CC + c] = acc;
}

// -------- map2token weighted scatter (onto downsampled grid) --------
__global__ void k_scatter2(const float* __restrict__ fmap2, const float* __restrict__ loc,
                           const int* __restrict__ idx_agg_t, const float* __restrict__ wgt_t,
                           float* __restrict__ m2t) {
    int token = blockIdx.x;                 // B*N blocks
    int b = token / NN;
    int c = threadIdx.x;
    int hw = grid_idx(loc[2 * token], loc[2 * token + 1], H2, W2);
    float feat = fmap2[((size_t)b * H2W2 + hw) * CC + c];
    int t = idx_agg_t[token];
    float wg = wgt_t[token];
    atomicAdd(&m2t[((size_t)b * NTOK + t) * CC + c], feat * wg);
}

// -------- combine skip+map paths, store xd, accumulate BN1 stats --------
__global__ void k_combine_bn1(const float* __restrict__ dd, const float* __restrict__ m2t,
                              const float* __restrict__ den, const float* __restrict__ skipw,
                              float* __restrict__ xd, float* __restrict__ bn1s,
                              float* __restrict__ bn1q) {
    const int RPB = MROWS / 256;            // 96 rows per block, 256 blocks
    int c = threadIdx.x;
    float sk = skipw[c];
    float s = 0.f, q = 0.f;
    for (int r = 0; r < RPB; ++r) {
        int row = blockIdx.x * RPB + r;
        float d = den[row] + 1e-6f;
        size_t o = (size_t)row * CC + c;
        // cover HBM latency on the two 25 MB streams (global_prefetch_b8)
        if (r + 4 < RPB) {
            __builtin_prefetch(&dd[o + 4 * CC], 0, 3);
            __builtin_prefetch(&m2t[o + 4 * CC], 0, 3);
        }
        float v = (dd[o] * sk + m2t[o]) / d;
        xd[o] = v;
        s += v;
        q += v * v;
    }
    atomicAdd(&bn1s[c], s);
    atomicAdd(&bn1q[c], q);
}

// -------- BN finalize: scale/bias from batch stats --------
__global__ void k_bnfin(const float* __restrict__ sum, const float* __restrict__ sumsq,
                        const float* __restrict__ gamma, const float* __restrict__ beta,
                        float* __restrict__ scale, float* __restrict__ bias, float invM) {
    int c = threadIdx.x;
    float mean = sum[c] * invM;
    float var = sumsq[c] * invM - mean * mean;
    float sc = gamma[c] * rsqrtf(var + 1e-5f);
    scale[c] = sc;
    bias[c] = beta[c] - mean * sc;
}

// -------- apply BN1 and pack activations to bf16 pairs --------
__global__ void k_bn1_to_bf16(const float* __restrict__ xd, const float* __restrict__ s1,
                              const float* __restrict__ b1, unsigned* __restrict__ xdbf) {
    size_t i = (size_t)blockIdx.x * blockDim.x + threadIdx.x;  // MROWS*CC/2 pairs
    int c0 = (int)((i % (CC / 2)) * 2);
    float v0 = xd[2 * i] * s1[c0] + b1[c0];
    float v1 = xd[2 * i + 1] * s1[c0 + 1] + b1[c0 + 1];
    xdbf[i] = pack_bf16(v0, v1);
}

// -------- WMMA bf16 GEMM: (24576 x 256) x (256 x 512) -> y f32 --------
// Each wave computes a 16(M) x 64(N) strip: the A fragment is reused across
// 4 N-tiles (4 WMMAs per K-chunk), raising intensity to 10 dwords/WMMA.
// Fragment layout per ISA 7.12.2 (16-bit A 16x32): lane half selects K+8,
// VGPR v holds packed K pair k0 = (v&4?16:0) + (v&3)*2 (+ half*8).
__global__ void k_gemm(const unsigned* __restrict__ xdbf, const unsigned* __restrict__ wbf,
                       float* __restrict__ y) {
    int lane = threadIdx.x & 31;
    int wave = threadIdx.x >> 5;
    int gw = blockIdx.x * 8 + wave;         // 12288 waves total
    int tn4 = gw & 7;                       // 8 N-strips of 64
    int tm = gw >> 3;                       // 1536 M-tiles
    int m0 = tm * 16, n0 = tn4 * 64;
    int half = lane >> 4, ml = lane & 15;

    const unsigned* arow = xdbf + (size_t)(m0 + ml) * (CC / 2);
    const unsigned* brow = wbf + (size_t)(n0 + ml) * (CC / 2);  // +2048 dwords per N-tile

    union Frag { v16bf v; unsigned u[8]; };
    v8f acc0 = {}, acc1 = {}, acc2 = {}, acc3 = {};
#pragma unroll
    for (int kc = 0; kc < CC / 32; ++kc) {  // 8 chunks of K=32
        Frag a, b0, b1, b2, b3;
#pragma unroll
        for (int v = 0; v < 8; ++v) {
            int kd = (kc * 32 + ((v & 4) ? 16 : 0) + (v & 3) * 2 + half * 8) >> 1;
            a.u[v]  = arow[kd];
            b0.u[v] = brow[kd];
            b1.u[v] = brow[kd + 16 * (CC / 2)];
            b2.u[v] = brow[kd + 32 * (CC / 2)];
            b3.u[v] = brow[kd + 48 * (CC / 2)];
        }
        acc0 = __builtin_amdgcn_wmma_f32_16x16x32_bf16(false, a.v, false, b0.v, (short)0, acc0, false, false);
        acc1 = __builtin_amdgcn_wmma_f32_16x16x32_bf16(false, a.v, false, b1.v, (short)0, acc1, false, false);
        acc2 = __builtin_amdgcn_wmma_f32_16x16x32_bf16(false, a.v, false, b2.v, (short)0, acc2, false, false);
        acc3 = __builtin_amdgcn_wmma_f32_16x16x32_bf16(false, a.v, false, b3.v, (short)0, acc3, false, false);
    }
    // C/D layout: VGPR r -> M = r + half*8, N = ml (+16 per sub-tile)
#pragma unroll
    for (int r = 0; r < 8; ++r) {
        size_t row = (size_t)(m0 + r + half * 8) * COUT;
        y[row + n0 + ml]      = acc0[r];
        y[row + n0 + 16 + ml] = acc1[r];
        y[row + n0 + 32 + ml] = acc2[r];
        y[row + n0 + 48 + ml] = acc3[r];
    }
}

// -------- BN2 stats over GEMM output --------
__global__ void k_bn2_stats(const float* __restrict__ y, float* __restrict__ bn2s,
                            float* __restrict__ bn2q) {
    const int RPB = MROWS / 256;            // 96 rows per block, 256 blocks, 512 threads
    int c = threadIdx.x;
    float s = 0.f, q = 0.f;
    for (int r = 0; r < RPB; ++r) {
        size_t row = (size_t)blockIdx.x * RPB + r;
        float v = y[row * COUT + c];
        s += v;
        q += v * v;
    }
    atomicAdd(&bn2s[c], s);
    atomicAdd(&bn2q[c], q);
}

// -------- BN2 apply + ReLU -> d_out --------
__global__ void k_out(const float* __restrict__ y, const float* __restrict__ s2,
                      const float* __restrict__ b2, float* __restrict__ out) {
    size_t total = (size_t)MROWS * COUT;
    size_t i = (size_t)blockIdx.x * blockDim.x + threadIdx.x;
    size_t st = (size_t)gridDim.x * blockDim.x;
    for (; i < total; i += st) {
        int c = (int)(i & (COUT - 1));
        float v = y[i] * s2[c] + b2[c];
        out[i] = v > 0.f ? v : 0.f;
    }
}

extern "C" void kernel_launch(void* const* d_in, const int* in_sizes, int n_in,
                              void* d_out, int out_size, void* d_ws, size_t ws_size,
                              hipStream_t stream) {
    const float* x       = (const float*)d_in[0];
    const float* loc     = (const float*)d_in[1];
    const int*   idxa    = (const int*)d_in[2];
    // d_in[3] agg_weight: unused by reference body
    // d_in[4] x_t: only shape used
    const int*   idxat   = (const int*)d_in[5];
    const float* wgtt    = (const float*)d_in[6];
    // d_in[7], d_in[8]: H, W scalars (compile-time constants here)
    const float* dwconv  = (const float*)d_in[9];   // (C,1,3,3)
    const float* dwskip  = (const float*)d_in[10];  // (C,1,1,1)
    const float* g1      = (const float*)d_in[11];
    const float* be1     = (const float*)d_in[12];
    const float* convw   = (const float*)d_in[13];  // (CO,C)
    const float* g2      = (const float*)d_in[14];
    const float* be2     = (const float*)d_in[15];
    float* out = (float*)d_out;

    float* ws = (float*)d_ws;
    float* map_sum = ws + OFF_MAPSUM;
    float* map_cnt = ws + OFF_MAPCNT;
    float* dd      = ws + OFF_DD;
    float* m2t     = ws + OFF_M2T;
    float* den     = ws + OFF_DEN;
    float* bn1s    = ws + OFF_BN1S;
    float* bn1q    = ws + OFF_BN1Q;
    float* bn2s    = ws + OFF_BN2S;
    float* bn2q    = ws + OFF_BN2Q;
    float* fmap2   = ws + OFF_FMAP2;
    unsigned* wbf  = (unsigned*)(ws + OFF_WBF);
    float* s1      = ws + OFF_S1;
    float* b1      = ws + OFF_B1;
    float* s2      = ws + OFF_S2;
    float* b2      = ws + OFF_B2;
    float* xd      = ws + OFF_XD;               // aliases dead map_sum
    float* y       = ws + OFF_Y;                // aliases dead map_sum
    unsigned* xdbf = (unsigned*)(ws + OFF_XDBF);// aliases dead map_sum

    // 1) zero accumulators
    k_zero<<<4096, 256, 0, stream>>>((float4*)ws, ZTOT / 4);
    // 2) weights -> bf16 (CO*C/2 = 65536 dwords)
    k_convert_w<<<256, 256, 0, stream>>>(convw, wbf);
    // 3) fused token2map + token_downup scatter
    k_scatter1<<<BB * NN, CC, 0, stream>>>(x, loc, idxa, idxat, wgtt,
                                           map_sum, map_cnt, dd, den);
    // 4) normalize + depthwise 3x3 stride-2
    k_dwconv<<<BB * H2W2, CC, 0, stream>>>(map_sum, map_cnt, dwconv, fmap2);
    // 5) map2token scatter
    k_scatter2<<<BB * NN, CC, 0, stream>>>(fmap2, loc, idxat, wgtt, m2t);
    // 6) combine + BN1 stats (map_sum dead; xd aliases it)
    k_combine_bn1<<<256, CC, 0, stream>>>(dd, m2t, den, dwskip, xd, bn1s, bn1q);
    // 7) BN1 finalize
    k_bnfin<<<1, CC, 0, stream>>>(bn1s, bn1q, g1, be1, s1, b1, 1.f / (float)MROWS);
    // 8) apply BN1, pack bf16 (MROWS*CC/2 = 3145728 pairs)
    k_bn1_to_bf16<<<12288, 256, 0, stream>>>(xd, s1, b1, xdbf);
    // 9) WMMA GEMM (12288 waves, each 16x64; 8 waves per block)
    k_gemm<<<1536, 256, 0, stream>>>(xdbf, wbf, y);
    // 10) BN2 stats
    k_bn2_stats<<<256, COUT, 0, stream>>>(y, bn2s, bn2q);
    // 11) BN2 finalize
    k_bnfin<<<1, COUT, 0, stream>>>(bn2s, bn2q, g2, be2, s2, b2, 1.f / (float)MROWS);
    // 12) BN2 apply + ReLU
    k_out<<<4096, 256, 0, stream>>>(y, s2, b2, out);
}
